// Transformer_77137612636563
// MI455X (gfx1250) — matrix operand order repaired and checked by reading, not compiled, not run on hardware
//
#include <hip/hip_runtime.h>
#include <hip/hip_bf16.h>

// ---------------------------------------------------------------------------
// MI455X (gfx1250, wave32) transformer layer.
// GEMMs on v_wmma_f32_16x16x32_f16 (f16 in, f32 accumulate). Global->LDS tile
// staging runs on the Tensor Data Mover with DOUBLE BUFFERING: TDM for tile
// k+1 is issued before the WMMAs on tile k, s_wait_tensorcnt only afterwards,
// so the DMA is hidden behind the matrix pipe.
// Shapes (static): B=8 L=1024 D=512 H=8 dh=64 F=1024, M = B*L = 8192.
// ---------------------------------------------------------------------------

typedef _Float16 half_t;
typedef __attribute__((ext_vector_type(16))) _Float16 v16h;
typedef __attribute__((ext_vector_type(8)))  _Float16 v8h;
typedef __attribute__((ext_vector_type(8)))  float    v8f;

#define LANE (threadIdx.x & 31)

// ---------------- TDM (Tensor Data Mover) support --------------------------
#if defined(__has_builtin)
#if __has_builtin(__builtin_amdgcn_tensor_load_to_lds)
#define HAVE_TDM 1
#endif
#endif

#if defined(__has_include)
#if __has_include(<hip/amd_detail/amd_gfx1250_TDM.h>)
#define TDM_6ARG 1   // therock-10.0 headers => clang-23 6-arg builtin
#endif
#endif

#ifdef HAVE_TDM
typedef __attribute__((ext_vector_type(4))) unsigned tdm_v4u;
typedef __attribute__((ext_vector_type(8))) int      tdm_v8i;
typedef __attribute__((ext_vector_type(4))) int      tdm_v4i;

// 2D tile load: tile_h rows of tile_w 2-byte elements, source rows strided by
// row_stride elements, packed contiguously into LDS. D# per CDNA5 ISA 8.3/8.4.
__device__ __forceinline__ void tdm_load_2d(unsigned lds_byte_off,
                                            const void* gptr,
                                            unsigned tile_w, unsigned tile_h,
                                            unsigned row_stride) {
  unsigned long long ga = (unsigned long long)gptr;
  tdm_v4u g0;
  g0[0] = 1u;                                                  // count=1
  g0[1] = lds_byte_off;                                        // lds_addr
  g0[2] = (unsigned)ga;                                        // global lo
  g0[3] = (unsigned)((ga >> 32) & 0x01FFFFFFu) | 0x80000000u;  // hi | type=2
  tdm_v8i g1;
  g1[0] = (int)(1u << 16);                                     // data_size=2B
  g1[1] = (int)((row_stride & 0xFFFFu) << 16);                 // tensor_dim0 lo
  g1[2] = (int)((row_stride >> 16) | ((tile_h & 0xFFFFu) << 16)); // d0 hi|d1 lo
  g1[3] = (int)((tile_w & 0xFFFFu) << 16);                     // d1 hi|tile_dim0
  g1[4] = (int)(tile_h & 0xFFFFu);                             // tile_dim1
  g1[5] = (int)row_stride;                                     // dim0_stride
  g1[6] = 0;
  g1[7] = 0;
  tdm_v4i z4 = {0, 0, 0, 0};
#ifdef TDM_6ARG
  tdm_v8i z8 = {0, 0, 0, 0, 0, 0, 0, 0};
  __builtin_amdgcn_tensor_load_to_lds(g0, g1, z4, z4, z8, 0);
#else
  __builtin_amdgcn_tensor_load_to_lds(g0, g1, z4, z4, 0);
#endif
}

__device__ __forceinline__ unsigned lds_off32(const void* p) {
  return (unsigned)(unsigned long long)p;  // low 32 bits = LDS byte offset
}
#endif  // HAVE_TDM

// ---------------- WMMA helpers ---------------------------------------------
__device__ __forceinline__ v8f wmma32(v16h a, v16h b, v8f c) {
  return __builtin_amdgcn_wmma_f32_16x16x32_f16(
      false, a, false, b, (short)0, c, false, false);
}

// A-fragment (16x32, 16-bit) from row-major global memory.
__device__ __forceinline__ v16h load_a16(const half_t* __restrict__ A, int lda,
                                         int row0, int k0) {
  int lane = LANE;
  int hi = lane >> 4, m = lane & 15;
  const half_t* p = A + (size_t)(row0 + m) * lda + k0 + 16 * hi;
  v8h lo = *(const v8h*)p;
  v8h h8 = *(const v8h*)(p + 8);
  v16h r;
#pragma unroll
  for (int i = 0; i < 8; ++i) { r[i] = lo[i]; r[i + 8] = h8[i]; }
  return r;
}

// B-fragment (32x16, 16-bit) from LDS laid out as bt[n][32 k-contig halves].
__device__ __forceinline__ v16h load_b16_lds(const half_t* bt, int n0) {
  int lane = LANE;
  int hi = lane >> 4, n = n0 + (lane & 15);
  return *(const v16h*)(bt + n * 32 + 16 * hi);
}

// ---------------------------------------------------------------------------
// Tiled WMMA GEMM:  C[M,N] = A[M,K](f16) @ B, B given pre-transposed Bt[N][K].
// Block tile 256(M) x 64(N), 8 waves x 32x64 (8 WMMA / k-step). B k-slab
// (64 x 32 halves) double-buffered in LDS: TDM for the next slab issued
// before the current slab's WMMAs, waited only afterwards.
// ---------------------------------------------------------------------------
__global__ __launch_bounds__(256) void gemm_wmma(
    const half_t* __restrict__ A, const half_t* __restrict__ Bt,
    const float* __restrict__ bias, half_t* __restrict__ C16,
    float* __restrict__ C32, int M, int N, int K, int relu) {
  __shared__ __align__(64) half_t bt[2][64 * 32];
  int tid = threadIdx.x;
  int lane = tid & 31, w = tid >> 5;
  int nblk = blockIdx.x * 64;
  int row0 = blockIdx.y * 256 + w * 32;

  v8f c0[4] = {v8f{}, v8f{}, v8f{}, v8f{}};
  v8f c1[4] = {v8f{}, v8f{}, v8f{}, v8f{}};

  // prologue: stage k0 = 0 into buffer 0
#ifdef HAVE_TDM
  if (w == 0) {
    tdm_load_2d(lds_off32(&bt[0][0]), Bt + (size_t)nblk * K,
                /*tile_w=*/32, /*tile_h=*/64, /*row_stride=*/(unsigned)K);
    __builtin_amdgcn_s_wait_tensorcnt(0);
  }
#else
  {
    int rr = tid >> 2, cx = (tid & 3) * 8;  // 64 rows x 32 halves
    v8h bv = *(const v8h*)(Bt + (size_t)(nblk + rr) * K + cx);
    *(v8h*)(&bt[0][rr * 32 + cx]) = bv;
  }
#endif
  __syncthreads();

  int cur = 0;
  for (int k0 = 0; k0 < K; k0 += 32) {
    int nxt = cur ^ 1;
    bool more = (k0 + 32) < K;

    // kick off staging of the NEXT slab (async; no wait yet)
#ifdef HAVE_TDM
    if (more && w == 0)
      tdm_load_2d(lds_off32(&bt[nxt][0]), Bt + (size_t)nblk * K + k0 + 32,
                  32, 64, (unsigned)K);
#else
    int rr = tid >> 2, cx = (tid & 3) * 8;
    v8h bnext = {};
    if (more)
      bnext = *(const v8h*)(Bt + (size_t)(nblk + rr) * K + k0 + 32 + cx);
#endif

    __builtin_prefetch(A + (size_t)(row0 + (lane & 15)) * K + k0 + 64, 0, 1);
    v16h a0 = load_a16(A, K, row0, k0);
    v16h a1 = load_a16(A, K, row0 + 16, k0);
#pragma unroll
    for (int ns = 0; ns < 4; ++ns) {
      v16h b = load_b16_lds(&bt[cur][0], ns * 16);
      c0[ns] = wmma32(a0, b, c0[ns]);
      c1[ns] = wmma32(a1, b, c1[ns]);
    }

    // complete next-slab staging, then block barrier
#ifdef HAVE_TDM
    if (more && w == 0) __builtin_amdgcn_s_wait_tensorcnt(0);
#else
    if (more) *(v8h*)(&bt[nxt][rr * 32 + cx]) = bnext;
#endif
    __syncthreads();
    cur = nxt;
  }

  int hi = lane >> 4, ln = lane & 15;
#pragma unroll
  for (int ns = 0; ns < 4; ++ns) {
    int col = nblk + ns * 16 + ln;
    float bb = bias[col];
#pragma unroll
    for (int r = 0; r < 8; ++r) {
      int rowA = row0 + r + 8 * hi;
      int rowB = rowA + 16;
      float va = c0[ns][r] + bb;
      float vb = c1[ns][r] + bb;
      if (relu) { va = fmaxf(va, 0.0f); vb = fmaxf(vb, 0.0f); }
      if (C32) {
        C32[(size_t)rowA * N + col] = va;
        C32[(size_t)rowB * N + col] = vb;
      }
      if (C16) {
        C16[(size_t)rowA * N + col] = (half_t)va;
        C16[(size_t)rowB * N + col] = (half_t)vb;
      }
    }
  }
}

// ---------------------------------------------------------------------------
// Attention on the torch-faithful contiguous head view [64][1024][64];
// mask row for head-batch bh is q_mask[bh % 8] (faithful tile() quirk).
// A = mask ? exp(S/8) : 0 ; denom = max(rowsum,1) ; ctx = (A@V)/denom.
// K tile via double-buffered TDM, V tile via register-pipelined transpose.
// ---------------------------------------------------------------------------
__global__ __launch_bounds__(256) void attn_wmma(
    const half_t* __restrict__ Q, const half_t* __restrict__ Kh,
    const half_t* __restrict__ V, const int* __restrict__ qmask,
    half_t* __restrict__ Ctx) {
  const int L = 1024, DH = 64;
  __shared__ __align__(64) half_t lds_k[2][32 * 64];   // [key_local][d]
  __shared__ __align__(64) half_t lds_v[2][64 * 32];   // [d][key_local]
  __shared__ __align__(64) half_t lds_p[8 * 16 * 32];  // per-wave P scratch

  int tid = threadIdx.x, lane = tid & 31, w = tid >> 5;
  int hi = lane >> 4, ln = lane & 15;
  int bh = blockIdx.y;
  int qrow0 = blockIdx.x * 128 + w * 16;

  const half_t* QH = Q + (size_t)bh * L * DH;
  const half_t* KH = Kh + (size_t)bh * L * DH;
  const half_t* VH = V + (size_t)bh * L * DH;
  const int* mrow = qmask + (size_t)(bh & 7) * L;
  half_t* pslot = lds_p + w * (16 * 32);

  int qmi[8];
#pragma unroll
  for (int r = 0; r < 8; ++r) qmi[r] = mrow[qrow0 + r + 8 * hi];

  // Q fragments are key-loop invariant: load once.
  v16h aq0 = load_a16(QH, DH, qrow0, 0);
  v16h aq1 = load_a16(QH, DH, qrow0, 32);

  float rs[8];
#pragma unroll
  for (int r = 0; r < 8; ++r) rs[r] = 0.0f;
  v8f cc[4] = {v8f{}, v8f{}, v8f{}, v8f{}};

  int kk = tid >> 3, cx = (tid & 7) * 8;

  // prologue: stage key chunk 0 into buffer 0
#ifdef HAVE_TDM
  if (w == 0)
    tdm_load_2d(lds_off32(&lds_k[0][0]), KH, /*tile_w=*/64, /*tile_h=*/32,
                /*row_stride=*/64);
#else
  {
    v8h kv = *(const v8h*)(KH + (size_t)kk * DH + cx);
    *(v8h*)(&lds_k[0][kk * 64 + cx]) = kv;
  }
#endif
  {
    v8h vv = *(const v8h*)(VH + (size_t)kk * DH + cx);
#pragma unroll
    for (int j = 0; j < 8; ++j) lds_v[0][(cx + j) * 32 + kk] = vv[j];
  }
#ifdef HAVE_TDM
  if (w == 0) __builtin_amdgcn_s_wait_tensorcnt(0);
#endif
  __syncthreads();

  int cur = 0;
  for (int kc = 0; kc < L; kc += 32) {
    int nxt = cur ^ 1;
    bool more = (kc + 32) < L;

    // kick off next chunk: TDM for K, registers for V (transposed later)
#ifdef HAVE_TDM
    if (more && w == 0)
      tdm_load_2d(lds_off32(&lds_k[nxt][0]), KH + (size_t)(kc + 32) * DH,
                  64, 32, 64);
#else
    v8h knext = {};
    if (more) knext = *(const v8h*)(KH + (size_t)(kc + 32 + kk) * DH + cx);
#endif
    v8h vnext = {};
    if (more) vnext = *(const v8h*)(VH + (size_t)(kc + 32 + kk) * DH + cx);

    // S (16 x 32 keys) = Q tile @ K^T : 2 k-steps over dh
    v8f s0 = {}, s1 = {};
#pragma unroll
    for (int d0 = 0; d0 < DH; d0 += 32) {
      v16h aq = d0 ? aq1 : aq0;
      v16h b0 = *(const v16h*)(&lds_k[cur][ln * 64 + d0 + 16 * hi]);
      v16h b1 = *(const v16h*)(&lds_k[cur][(16 + ln) * 64 + d0 + 16 * hi]);
      s0 = wmma32(aq, b0, s0);
      s1 = wmma32(aq, b1, s1);
    }

    // mask + exp, accumulate row sums, spill P row-major to per-wave LDS
#pragma unroll
    for (int ks = 0; ks < 2; ++ks) {
      int kmv = mrow[kc + ks * 16 + ln];
#pragma unroll
      for (int r = 0; r < 8; ++r) {
        float sv = (ks ? s1[r] : s0[r]) * 0.125f;  // 1/sqrt(64)
        float p = (qmi[r] != 0 && kmv != 0) ? __expf(sv) : 0.0f;
        rs[r] += p;
        pslot[(r + 8 * hi) * 32 + ks * 16 + ln] = (half_t)p;
      }
    }

    // P as A-fragment (16x32 over the 32 keys)
    v16h ap;
    {
      const half_t* pp = pslot + ln * 32;
      v8h plo = *(const v8h*)(pp + 16 * hi);
      v8h phi = *(const v8h*)(pp + 16 * hi + 8);
#pragma unroll
      for (int i = 0; i < 8; ++i) { ap[i] = plo[i]; ap[i + 8] = phi[i]; }
    }
#pragma unroll
    for (int ns = 0; ns < 4; ++ns) {
      v16h bv2 = *(const v16h*)(&lds_v[cur][(ns * 16 + ln) * 32 + 16 * hi]);
      cc[ns] = wmma32(ap, bv2, cc[ns]);
    }

    // finish staging of next chunk, then barrier
    if (more) {
#pragma unroll
      for (int j = 0; j < 8; ++j) lds_v[nxt][(cx + j) * 32 + kk] = vnext[j];
    }
#ifdef HAVE_TDM
    if (more && w == 0) __builtin_amdgcn_s_wait_tensorcnt(0);
#else
    if (more) *(v8h*)(&lds_k[nxt][kk * 64 + cx]) = knext;
#endif
    __syncthreads();
    cur = nxt;
  }

#pragma unroll
  for (int r = 0; r < 8; ++r) {
    float t = rs[r];
    t += __shfl_xor(t, 1);
    t += __shfl_xor(t, 2);
    t += __shfl_xor(t, 4);
    t += __shfl_xor(t, 8);
    float inv = 1.0f / fmaxf(t, 1.0f);
    size_t row = (size_t)bh * L + qrow0 + r + 8 * hi;
#pragma unroll
    for (int ns = 0; ns < 4; ++ns)
      Ctx[row * DH + ns * 16 + ln] = (half_t)(cc[ns][r] * inv);
  }
}

// ---------------------------------------------------------------------------
// Fused residual-add + LayerNorm over D=512, one row per block (256 thr).
// ---------------------------------------------------------------------------
__global__ __launch_bounds__(256) void ln_kernel(
    const float* __restrict__ x, const float* __restrict__ res,
    const float* __restrict__ g, const float* __restrict__ b,
    float* __restrict__ o32, half_t* __restrict__ o16) {
  __shared__ float sred[256];
  int tid = threadIdx.x;
  size_t base = (size_t)blockIdx.x * 512;
  float v0 = x[base + tid] + res[base + tid];
  float v1 = x[base + tid + 256] + res[base + tid + 256];

  sred[tid] = v0 + v1;
  __syncthreads();
  for (int st = 128; st > 0; st >>= 1) {
    if (tid < st) sred[tid] += sred[tid + st];
    __syncthreads();
  }
  float mu = sred[0] * (1.0f / 512.0f);
  __syncthreads();

  float d0 = v0 - mu, d1 = v1 - mu;
  sred[tid] = d0 * d0 + d1 * d1;
  __syncthreads();
  for (int st = 128; st > 0; st >>= 1) {
    if (tid < st) sred[tid] += sred[tid + st];
    __syncthreads();
  }
  float rstd = rsqrtf(sred[0] * (1.0f / 512.0f) + 1e-5f);

  float y0 = d0 * rstd * g[tid] + b[tid];
  float y1 = d1 * rstd * g[tid + 256] + b[tid + 256];
  o32[base + tid] = y0;
  o32[base + tid + 256] = y1;
  if (o16) {
    o16[base + tid] = (half_t)y0;
    o16[base + tid + 256] = (half_t)y1;
  }
}

__global__ void cvt_f32_f16(const float* __restrict__ in,
                            half_t* __restrict__ out, int n) {
  int i = blockIdx.x * blockDim.x + threadIdx.x;
  int stride = gridDim.x * blockDim.x;
  for (; i < n; i += stride) out[i] = (half_t)in[i];
}

// weight convert + transpose: in[K][N] f32 -> out[N][K] f16
__global__ void cvt_t_f16(const float* __restrict__ in,
                          half_t* __restrict__ out, int K, int N) {
  int idx = blockIdx.x * blockDim.x + threadIdx.x;
  int total = K * N;
  int stride = gridDim.x * blockDim.x;
  for (; idx < total; idx += stride) {
    int n = idx / K, k = idx - n * K;
    out[idx] = (half_t)in[(size_t)k * N + n];
  }
}

// outputs 2..5 of the tuple: q_mask (as float), query, query, q_mask
__global__ void pack_tail(const float* __restrict__ query,
                          const int* __restrict__ qmask,
                          float* __restrict__ out) {
  const size_t NQ = (size_t)8192 * 512;
  const size_t NM = 8192;
  const size_t o1 = NQ, o2 = NQ + NM, o3 = o2 + NQ, o4 = o3 + NQ;
  size_t i = (size_t)blockIdx.x * blockDim.x + threadIdx.x;
  size_t stride = (size_t)gridDim.x * blockDim.x;
  for (size_t j = i; j < NQ; j += stride) {
    float v = query[j];
    out[o2 + j] = v;
    out[o3 + j] = v;
  }
  for (size_t j = i; j < NM; j += stride) {
    float mv = (float)qmask[j];
    out[o1 + j] = mv;
    out[o4 + j] = mv;
  }
}

// ---------------------------------------------------------------------------
extern "C" void kernel_launch(void* const* d_in, const int* in_sizes, int n_in,
                              void* d_out, int out_size, void* d_ws,
                              size_t ws_size, hipStream_t stream) {
  (void)in_sizes; (void)n_in; (void)out_size; (void)ws_size;
  const int M = 8192, D = 512, F = 1024;

  const float* query = (const float*)d_in[0];
  const int*   qmask = (const int*)d_in[1];
  const float* wq = (const float*)d_in[2];
  const float* bq = (const float*)d_in[3];
  const float* wk = (const float*)d_in[4];
  const float* bk = (const float*)d_in[5];
  const float* wv = (const float*)d_in[6];
  const float* bv = (const float*)d_in[7];
  const float* wo = (const float*)d_in[8];
  const float* bo = (const float*)d_in[9];
  const float* ln1g = (const float*)d_in[10];
  const float* ln1b = (const float*)d_in[11];
  const float* w1 = (const float*)d_in[12];
  const float* b1 = (const float*)d_in[13];
  const float* w2 = (const float*)d_in[14];
  const float* b2 = (const float*)d_in[15];
  const float* ln2g = (const float*)d_in[16];
  const float* ln2b = (const float*)d_in[17];

  char* ws = (char*)d_ws;
  size_t off = 0;
  auto alloc = [&](size_t bytes) {
    char* p = ws + off;
    off += (bytes + 255) & ~(size_t)255;
    return p;
  };
  half_t* Xh   = (half_t*)alloc((size_t)M * D * 2);
  half_t* WqT  = (half_t*)alloc((size_t)D * D * 2);   // [N][K]
  half_t* WkT  = (half_t*)alloc((size_t)D * D * 2);
  half_t* WvT  = (half_t*)alloc((size_t)D * D * 2);
  half_t* WoT  = (half_t*)alloc((size_t)D * D * 2);
  half_t* W1T  = (half_t*)alloc((size_t)D * F * 2);   // [F][D]
  half_t* W2T  = (half_t*)alloc((size_t)F * D * 2);   // [D][F]
  half_t* Qh   = (half_t*)alloc((size_t)M * D * 2);
  half_t* Kh   = (half_t*)alloc((size_t)M * D * 2);
  half_t* Vh   = (half_t*)alloc((size_t)M * D * 2);
  half_t* Ctxh = (half_t*)alloc((size_t)M * D * 2);
  float*  attnout = (float*)alloc((size_t)M * D * 4);
  float*  out1    = (float*)alloc((size_t)M * D * 4);
  half_t* out1h   = (half_t*)alloc((size_t)M * D * 2);
  half_t* ffnh    = (half_t*)alloc((size_t)M * F * 2);
  float*  ffnout  = (float*)alloc((size_t)M * D * 4);

  // 1) conversions (weights transposed once; reused across all 32 M-blocks)
  cvt_f32_f16<<<1024, 256, 0, stream>>>(query, Xh, M * D);
  cvt_t_f16<<<256, 256, 0, stream>>>(wq, WqT, D, D);
  cvt_t_f16<<<256, 256, 0, stream>>>(wk, WkT, D, D);
  cvt_t_f16<<<256, 256, 0, stream>>>(wv, WvT, D, D);
  cvt_t_f16<<<256, 256, 0, stream>>>(wo, WoT, D, D);
  cvt_t_f16<<<512, 256, 0, stream>>>(w1, W1T, D, F);
  cvt_t_f16<<<512, 256, 0, stream>>>(w2, W2T, F, D);

  // 2) QKV projections
  dim3 gD(D / 64, M / 256);   // (8, 32)
  dim3 gF(F / 64, M / 256);   // (16, 32)
  gemm_wmma<<<gD, 256, 0, stream>>>(Xh, WqT, bq, Qh, nullptr, M, D, D, 0);
  gemm_wmma<<<gD, 256, 0, stream>>>(Xh, WkT, bk, Kh, nullptr, M, D, D, 0);
  gemm_wmma<<<gD, 256, 0, stream>>>(Xh, WvT, bv, Vh, nullptr, M, D, D, 0);

  // 3) attention over the contiguous [64][1024][64] head view
  attn_wmma<<<dim3(8, 64), 256, 0, stream>>>(Qh, Kh, Vh, qmask, Ctxh);

  // 4) out-projection
  gemm_wmma<<<gD, 256, 0, stream>>>(Ctxh, WoT, bo, nullptr, attnout, M, D, D, 0);

  // 5) out = LN(query + attnout)
  ln_kernel<<<M, 256, 0, stream>>>(attnout, query, ln1g, ln1b, out1, out1h);

  // 6) FFN
  gemm_wmma<<<gF, 256, 0, stream>>>(out1h, W1T, b1, ffnh, nullptr, M, F, D, 1);
  gemm_wmma<<<gD, 256, 0, stream>>>(ffnh, W2T, b2, nullptr, ffnout, M, D, F, 0);

  // 7) out2 = LN(out + ffn) -> first tuple member in d_out
  ln_kernel<<<M, 256, 0, stream>>>(ffnout, out1, ln2g, ln2b, (float*)d_out,
                                   nullptr);

  // 8) remaining tuple members
  pack_tail<<<2048, 256, 0, stream>>>(query, qmask, (float*)d_out);
}
